// MosaicMemory_7138235646544
// MI455X (gfx1250) — matrix operand order repaired and checked by reading, not compiled, not run on hardware
//
#include <hip/hip_runtime.h>
#include <hip/hip_bf16.h>
#include <math.h>

// ---------------- Problem constants (match reference) ----------------
#define BB   4
#define SS   4096
#define DD   1024
#define NN   (BB * SS)        // 16384 tokens
#define KK   32               // mem_key_dim
#define HH   4                // hashes
#define AA   2                // assoc
#define TT   65536            // buckets
#define NBITS 16
#define DM   64               // mem_dim
#define TV   32               // vsa dim

typedef __bf16 bf16_t;
typedef __bf16 v16bf __attribute__((ext_vector_type(16)));
typedef float  v8f   __attribute__((ext_vector_type(8)));

// ---------------- helpers -------------------------------------------
__device__ __forceinline__ unsigned pack2bf(float a, float b) {
  union { unsigned u; __bf16 h[2]; } p;
  p.h[0] = (__bf16)a; p.h[1] = (__bf16)b;
  return p.u;
}

__device__ __forceinline__ float wave_sum(float x) {
#pragma unroll
  for (int o = 16; o > 0; o >>= 1) x += __shfl_xor(x, o, 32);
  return x;
}

// Async global->LDS copy, one DWORD per lane (ASYNCcnt-tracked).
__device__ __forceinline__ void async_gather_b32(void* lds_dst, const float* gptr) {
  const unsigned lds_off = (unsigned)(uintptr_t)lds_dst;         // low 32b = LDS offset
  asm volatile("global_load_async_to_lds_b32 %0, %1, off"
               :: "v"(lds_off), "v"((unsigned long long)(uintptr_t)gptr)
               : "memory");
}
__device__ __forceinline__ void async_wait_all() {
  asm volatile("s_wait_asynccnt 0x0" ::: "memory");
}

// ---------------- WMMA fragment loaders (wave32, 16x16x32 bf16) ------
// A (16x32, MxK): lane L -> M = L%16; half = L/16.
// VGPR v holds K = (v>=4?16:0) + half*8 + (v%4)*2 + {0,1}
__device__ __forceinline__ v16bf load_a_frag(const bf16_t* sm, int row0, int ld, int lane) {
  const int m = lane & 15, half = lane >> 4;
  v16bf a;
#pragma unroll
  for (int e = 0; e < 16; ++e) {
    const int v = e >> 1, p = e & 1;
    const int k = ((v & 4) ? 16 : 0) + half * 8 + ((v & 3) << 1) + p;
    a[e] = sm[(row0 + m) * ld + k];
  }
  return a;
}
// B (32x16, KxN): lane L -> N = L%16; half = L/16; element e holds
// K = (e>=8?16:0) + half*8 + e%8. sm stores B column-major: sm[n*ld + k]
__device__ __forceinline__ v16bf load_b_frag(const bf16_t* sm, int col0, int ld, int lane) {
  const int n = lane & 15, half = lane >> 4;
  v16bf b;
#pragma unroll
  for (int e = 0; e < 16; ++e) {
    const int k = ((e & 8) << 1) + half * 8 + (e & 7);
    b[e] = sm[(col0 + n) * ld + k];
  }
  return b;
}

// =====================================================================
// Kernel 1: fused projection GEMM, double-buffered LDS pipeline.
//   C[N,128] = U[N,1024] x Wcat^T  (cols 0-31 Wq, 32-63 Ww, 64-127 Wv)
//   plus g = sigmoid(u . Wg + bg) using the staged U tile.
// 256 threads = 8 waves; block computes 64 rows x 128 cols.
// =====================================================================
__device__ __forceinline__ void stage_proj_tiles(
    bf16_t* __restrict__ sU, bf16_t* __restrict__ sW, float* __restrict__ sWg,
    const float* __restrict__ u,
    const float* __restrict__ Wq, const float* __restrict__ Ww,
    const float* __restrict__ Wv, const float* __restrict__ Wg,
    int m_base, int kt, int tid)
{
  // U tile: 64 rows x 32 K = 512 float4 loads (2/thread), b64 LDS stores
#pragma unroll
  for (int i = 0; i < 2; ++i) {
    const int e4 = i * 256 + tid;
    const int r = e4 >> 3, c4 = e4 & 7;
    const float4 f = *(const float4*)(u + (size_t)(m_base + r) * DD + kt + c4 * 4);
    uint2 pk; pk.x = pack2bf(f.x, f.y); pk.y = pack2bf(f.z, f.w);
    *(uint2*)(&sU[r * 32 + c4 * 4]) = pk;
  }
  // Wcat tile: 128 cols x 32 K = 1024 float4 loads (4/thread)
#pragma unroll
  for (int i = 0; i < 4; ++i) {
    const int e4 = i * 256 + tid;
    const int n = e4 >> 3, c4 = e4 & 7;
    const float* src = (n < 32) ? (Wq + (size_t)n * DD)
                     : (n < 64) ? (Ww + (size_t)(n - 32) * DD)
                                : (Wv + (size_t)(n - 64) * DD);
    const float4 f = *(const float4*)(src + kt + c4 * 4);
    uint2 pk; pk.x = pack2bf(f.x, f.y); pk.y = pack2bf(f.z, f.w);
    *(uint2*)(&sW[n * 32 + c4 * 4]) = pk;
  }
  if (tid < 32) sWg[tid] = Wg[kt + tid];
}

__global__ __launch_bounds__(256) void proj_kernel(
    const float* __restrict__ u,
    const float* __restrict__ Wq, const float* __restrict__ Ww,
    const float* __restrict__ Wv, const float* __restrict__ Wg,
    const float* __restrict__ bg,
    float* __restrict__ q_out, float* __restrict__ wk_out,
    float* __restrict__ v_out, float* __restrict__ g_out)
{
  __shared__ bf16_t sU[2][64 * 32];    // ping-pong U tiles (8 KB)
  __shared__ bf16_t sW[2][128 * 32];   // ping-pong W tiles (16 KB)
  __shared__ float  sWg[2][32];

  const int tid  = threadIdx.x;
  const int lane = tid & 31, wave = tid >> 5;
  const int m_base = blockIdx.x * 64;
  const int n0 = wave * 16;

  v8f acc[4];
#pragma unroll
  for (int s = 0; s < 4; ++s) acc[s] = (v8f){0.f,0.f,0.f,0.f,0.f,0.f,0.f,0.f};
  float ga = 0.f;

  stage_proj_tiles(sU[0], sW[0], sWg[0], u, Wq, Ww, Wv, Wg, m_base, 0, tid);
  __syncthreads();

  int buf = 0;
  for (int kt = 0; kt < DD; kt += 32) {
    const int nxt = buf ^ 1;
    // issue next tile's global loads; they drain while we run WMMAs below
    if (kt + 32 < DD)
      stage_proj_tiles(sU[nxt], sW[nxt], sWg[nxt], u, Wq, Ww, Wv, Wg,
                       m_base, kt + 32, tid);

    // gate-dot partial (rows handled by tid 0..63)
    if (tid < 64) {
#pragma unroll
      for (int c = 0; c < 32; ++c)
        ga += (float)sU[buf][tid * 32 + c] * sWg[buf][c];
    }

    const v16bf b = load_b_frag(sW[buf], n0, 32, lane);
#pragma unroll
    for (int s = 0; s < 4; ++s) {
      const v16bf a = load_a_frag(sU[buf], s * 16, 32, lane);
      acc[s] = __builtin_amdgcn_wmma_f32_16x16x32_bf16(
          false, a, false, b, (short)0, acc[s], false, false);
    }
    __syncthreads();   // publishes nxt, retires buf
    buf = nxt;
  }

  // ---- epilogue: route columns to q / wk / v (wave-uniform branch) ----
  const int ncol = n0 + (lane & 15);
#pragma unroll
  for (int s = 0; s < 4; ++s) {
#pragma unroll
    for (int r = 0; r < 8; ++r) {
      const int m = m_base + s * 16 + (lane >> 4) * 8 + r;
      const float val = acc[s][r];
      if (ncol < 32)      q_out [(size_t)m * KK + ncol]        = val;
      else if (ncol < 64) wk_out[(size_t)m * KK + (ncol - 32)] = val;
      else                v_out [(size_t)m * DM + (ncol - 64)] = val;
    }
  }
  if (tid < 64) g_out[m_base + tid] = 1.f / (1.f + __expf(-(ga + bg[0])));
}

// =====================================================================
// Kernel 2: routing + read + novelty-gated scatter. One wave per token.
// Slot gathers staged via async global->LDS (ASYNCcnt) in one burst.
// =====================================================================
__global__ __launch_bounds__(256) void route_kernel(
    const float* __restrict__ q,  const float* __restrict__ wk,
    const float* __restrict__ v,  const float* __restrict__ g,
    const float* __restrict__ R,  const float* __restrict__ P,
    const float* __restrict__ mem_keys, const float* __restrict__ mem_vals,
    const float* __restrict__ mem_tags,
    float* __restrict__ read_out,
    float* __restrict__ out_keys, float* __restrict__ out_vals,
    float* __restrict__ out_tags)
{
  const int tid = threadIdx.x, lane = tid & 31, wave = tid >> 5;
  const int token = blockIdx.x * 8 + wave;

  __shared__ float sq [8][32], swk[8][32], sqt[8][32], swt[8][32];
  __shared__ int   sidx[8][4];
  __shared__ float ssc[8][8], smatch[8][8];
  __shared__ float sKg[8][8][32];   // [wave][slot][k]   staged keys
  __shared__ float sTg[8][8][32];   // staged tags
  __shared__ float sVg[8][8][64];   // staged values

  const float ql  = q [(size_t)token * KK + lane];
  const float wkl = wk[(size_t)token * KK + lane];
  sq [wave][lane] = ql;
  swk[wave][lane] = wkl;
  if (lane < HH) sidx[wave][lane] = 0;

  // VSA tags of query / write-key: lane t computes one tag element
  float aq = 0.f, aw = 0.f;
#pragma unroll
  for (int k = 0; k < KK; ++k) {
    const float pk = P[lane * KK + k];
    aq += sq [wave][k] * pk;
    aw += swk[wave][k] * pk;
  }
  const float qtl = tanhf(aq), wtl = tanhf(aw);
  sqt[wave][lane] = qtl;
  swt[wave][lane] = wtl;

  // bit-router: 64 sign bits (H=4 x NB=16), two per-lane passes
#pragma unroll
  for (int p = 0; p < 2; ++p) {
    const int hn = p * 32 + lane;
    const int h = hn >> 4, nb = hn & 15;
    float d = 0.f;
#pragma unroll
    for (int k = 0; k < KK; ++k) d += sq[wave][k] * R[(h * KK + k) * NBITS + nb];
    if (d > 0.f) atomicOr(&sidx[wave][h], 1 << nb);
  }

  // ---- burst all 8 slots' gathers as async global->LDS transfers ----
#pragma unroll
  for (int s = 0; s < HH * AA; ++s) {
    const int h = s >> 1, a = s & 1;
    const size_t slot = (size_t)sidx[wave][h] * AA + a;
    async_gather_b32(&sKg[wave][s][lane], mem_keys + slot * KK + lane);
    async_gather_b32(&sTg[wave][s][lane], mem_tags + slot * TV + lane);
    async_gather_b32(&sVg[wave][s][lane],      mem_vals + slot * DM + lane);
    async_gather_b32(&sVg[wave][s][32 + lane], mem_vals + slot * DM + 32 + lane);
  }
  async_wait_all();   // drain ASYNCcnt for this wave

  // ---- scores + match via wave reductions over the staged data ----
  const float INV_SQRT_K = 0.17677669529663689f;  // 1/sqrt(32)
#pragma unroll
  for (int s = 0; s < HH * AA; ++s) {
    const float kg = sKg[wave][s][lane];
    const float tg = sTg[wave][s][lane];
    const float sc = wave_sum(ql * kg) * INV_SQRT_K
                   + wave_sum(qtl * tg) * (1.f / TV);     // VSA_W=1
    const float mt = wave_sum(wtl * tg) * (1.f / TV);
    if (lane == 0) { ssc[wave][s] = sc; smatch[wave][s] = mt; }
  }

  // softmax over 8 slots (READ_TEMP = 1)
  float mx = -1e30f;
#pragma unroll
  for (int s = 0; s < 8; ++s) mx = fmaxf(mx, ssc[wave][s]);
  float at[8], denom = 0.f;
#pragma unroll
  for (int s = 0; s < 8; ++s) { at[s] = __expf(ssc[wave][s] - mx); denom += at[s]; }
  const float inv = 1.f / denom;

  // read vector: lane covers d = lane and d = lane+32
  float r0 = 0.f, r1 = 0.f;
#pragma unroll
  for (int s = 0; s < 8; ++s) {
    const float w = at[s] * inv;
    r0 += w * sVg[wave][s][lane];
    r1 += w * sVg[wave][s][32 + lane];
  }
  read_out[(size_t)token * DM + lane]      = r0;
  read_out[(size_t)token * DM + 32 + lane] = r1;

  // write path: gate, per-hash argmax slot, novelty, scatter deltas.
  // "cur" comes from the LDS-staged ORIGINAL table entries (= reference).
  const float gv   = g[token];
  const float gate = (gv > 0.5f) ? gv : 0.f;             // g * (g > WRITE_THR)
  const float v0 = v[(size_t)token * DM + lane];
  const float v1 = v[(size_t)token * DM + 32 + lane];
#pragma unroll
  for (int h = 0; h < HH; ++h) {
    const float m0 = smatch[wave][h * 2], m1 = smatch[wave][h * 2 + 1];
    const int   sl = (m1 > m0) ? 1 : 0;                  // argmax (first on tie)
    const int   s  = h * 2 + sl;
    const float mmax = fmaxf(m0, m1);
    const float nov  = 1.f / (1.f + __expf(mmax));       // sigmoid(BETA*(0 - max))
    const float w    = 0.1f * gate * nov;                // ETA = 0.1
    const size_t fs  = (size_t)sidx[wave][h] * AA + sl;
    atomicAdd(&out_keys[fs * KK + lane], w * (swk[wave][lane] - sKg[wave][s][lane]));
    atomicAdd(&out_tags[fs * TV + lane], w * (swt[wave][lane] - sTg[wave][s][lane]));
    atomicAdd(&out_vals[fs * DM + lane],      w * (v0 - sVg[wave][s][lane]));
    atomicAdd(&out_vals[fs * DM + 32 + lane], w * (v1 - sVg[wave][s][32 + lane]));
  }
}

// =====================================================================
// Kernel 3: y[N,1024] = read[N,64] x Wout[1024,64]^T
// K=64 staged entirely in LDS once; 8 back-to-back WMMAs, one barrier.
// =====================================================================
__global__ __launch_bounds__(256) void ygemm_kernel(
    const float* __restrict__ readb, const float* __restrict__ Wout,
    float* __restrict__ y)
{
  __shared__ bf16_t sR[64 * 64];     // 8 KB
  __shared__ bf16_t sWo[128 * 64];   // 16 KB
  const int tid = threadIdx.x, lane = tid & 31, wave = tid >> 5;
  const int m_base = blockIdx.x * 64;
  const int d_base = blockIdx.y * 128;
  const int n0 = wave * 16;

  // stage read tile: 64 rows x 64 K = 1024 float4 (4/thread)
#pragma unroll
  for (int i = 0; i < 4; ++i) {
    const int e4 = i * 256 + tid;
    const int r = e4 >> 4, c4 = e4 & 15;
    const float4 f = *(const float4*)(readb + (size_t)(m_base + r) * DM + c4 * 4);
    uint2 pk; pk.x = pack2bf(f.x, f.y); pk.y = pack2bf(f.z, f.w);
    *(uint2*)(&sR[r * 64 + c4 * 4]) = pk;
  }
  // stage Wout tile: 128 cols x 64 K = 2048 float4 (8/thread)
#pragma unroll
  for (int i = 0; i < 8; ++i) {
    const int e4 = i * 256 + tid;
    const int n = e4 >> 4, c4 = e4 & 15;
    const float4 f = *(const float4*)(Wout + (size_t)(d_base + n) * DM + c4 * 4);
    uint2 pk; pk.x = pack2bf(f.x, f.y); pk.y = pack2bf(f.z, f.w);
    *(uint2*)(&sWo[n * 64 + c4 * 4]) = pk;
  }
  __syncthreads();

  v8f acc[4];
#pragma unroll
  for (int s = 0; s < 4; ++s) acc[s] = (v8f){0.f,0.f,0.f,0.f,0.f,0.f,0.f,0.f};

#pragma unroll
  for (int kt = 0; kt < DM; kt += 32) {
    const v16bf b = load_b_frag(sWo + kt, n0, 64, lane);
#pragma unroll
    for (int s = 0; s < 4; ++s) {
      const v16bf a = load_a_frag(sR + kt, s * 16, 64, lane);
      acc[s] = __builtin_amdgcn_wmma_f32_16x16x32_bf16(
          false, a, false, b, (short)0, acc[s], false, false);
    }
  }

  const int ncol = d_base + n0 + (lane & 15);
#pragma unroll
  for (int s = 0; s < 4; ++s) {
#pragma unroll
    for (int r = 0; r < 8; ++r) {
      const int m = m_base + s * 16 + (lane >> 4) * 8 + r;
      y[(size_t)m * DD + ncol] = acc[s][r];
    }
  }
}

// =====================================================================
extern "C" void kernel_launch(void* const* d_in, const int* in_sizes, int n_in,
                              void* d_out, int out_size, void* d_ws, size_t ws_size,
                              hipStream_t stream) {
  const float* u        = (const float*)d_in[0];
  const float* Wq       = (const float*)d_in[1];
  const float* Ww       = (const float*)d_in[2];
  const float* Wv       = (const float*)d_in[3];
  const float* Wout     = (const float*)d_in[4];
  const float* Wg       = (const float*)d_in[5];
  const float* bg       = (const float*)d_in[6];
  const float* R        = (const float*)d_in[7];
  const float* P        = (const float*)d_in[8];
  const float* mem_keys = (const float*)d_in[9];
  const float* mem_vals = (const float*)d_in[10];
  const float* mem_tags = (const float*)d_in[11];

  float* y        = (float*)d_out;
  float* out_keys = y + (size_t)NN * DD;
  float* out_vals = out_keys + (size_t)TT * AA * KK;
  float* out_tags = out_vals + (size_t)TT * AA * DM;

  float* ws_q    = (float*)d_ws;
  float* ws_wk   = ws_q    + (size_t)NN * KK;
  float* ws_v    = ws_wk   + (size_t)NN * KK;
  float* ws_g    = ws_v    + (size_t)NN * DM;
  float* ws_read = ws_g    + (size_t)NN;

  // seed output tables with original memory (deltas scatter-added on top)
  hipMemcpyAsync(out_keys, mem_keys, sizeof(float) * (size_t)TT * AA * KK,
                 hipMemcpyDeviceToDevice, stream);
  hipMemcpyAsync(out_vals, mem_vals, sizeof(float) * (size_t)TT * AA * DM,
                 hipMemcpyDeviceToDevice, stream);
  hipMemcpyAsync(out_tags, mem_tags, sizeof(float) * (size_t)TT * AA * TV,
                 hipMemcpyDeviceToDevice, stream);

  proj_kernel<<<NN / 64, 256, 0, stream>>>(u, Wq, Ww, Wv, Wg, bg,
                                           ws_q, ws_wk, ws_v, ws_g);
  route_kernel<<<NN / 8, 256, 0, stream>>>(ws_q, ws_wk, ws_v, ws_g, R, P,
                                           mem_keys, mem_vals, mem_tags,
                                           ws_read, out_keys, out_vals, out_tags);
  dim3 yg(NN / 64, DD / 128);
  ygemm_kernel<<<yg, 256, 0, stream>>>(ws_read, Wout, y);
}
